// MultiHeadAttention_48301202211104
// MI455X (gfx1250) — matrix operand order repaired and checked by reading, not compiled, not run on hardware
//
#include <hip/hip_runtime.h>

#define D_MODEL 2048
#define NHEADS  16
#define HDIM    128
#define BATCH   2
#define SEQ     2048
#define MTOT    (BATCH*SEQ)

typedef __attribute__((ext_vector_type(16))) _Float16 v16h;
typedef __attribute__((ext_vector_type(8)))  float    v8f;

union HFrag { v16h v; unsigned u[8]; _Float16 h[16]; };

__device__ inline v8f wmma_f16(v16h a, v16h b, v8f c) {
    // D = A(16x32 f16) * B(32x16 f16) + C(16x16 f32)
    return __builtin_amdgcn_wmma_f32_16x16x32_f16(
        false, a, false, b, (short)0, c, false, false);
}

__device__ inline v8f v8f_zero() {
    v8f z = {0.f,0.f,0.f,0.f,0.f,0.f,0.f,0.f};
    return z;
}

// gfx1250 async global->LDS DMA: 16 bytes per lane, tracked by ASYNCcnt.
// lds_off = LDS byte address (low 32 bits of generic shared pointer),
// gsrc    = per-lane 64-bit global address.
__device__ inline void async_copy_b128(unsigned lds_off, const void* gsrc) {
    asm volatile("global_load_async_to_lds_b128 %0, %1, off"
                 :: "v"(lds_off), "v"(gsrc) : "memory");
}

__device__ inline void wait_asynccnt0() {
#if __has_builtin(__builtin_amdgcn_s_wait_asynccnt)
    __builtin_amdgcn_s_wait_asynccnt(0);
#else
    asm volatile("s_wait_asynccnt 0x0" ::: "memory");
#endif
}

// ---------------------------------------------------------------------------
// GEMM: Y[m,n] = sum_k A[m,k] * W[n,k] + bias[n]   (M=4096, N=K=2048)
// Optional RoPE epilogue; writes fp32 (d_out) and/or f16 (workspace).
// Block: 128x128 tile, 256 threads = 8 waves arranged 4(M) x 2(N);
// each wave computes 32x64 = 2x4 WMMA 16x16 tiles, K staged in LDS as f16.
// A_F16 (template): A already f16 -> stage lA with async global->LDS DMA.
// ---------------------------------------------------------------------------
template<int A_F16>
__global__ __launch_bounds__(256)
void gemm_rope_kernel(const void* __restrict__ Ain,
                      const float* __restrict__ W, const float* __restrict__ bias,
                      const float* __restrict__ cosT, const float* __restrict__ sinT,
                      int rope, float* __restrict__ out32,
                      _Float16* __restrict__ out16, int out_layout)
{
    __shared__ _Float16 lA[128*32];  // [m][k]
    __shared__ _Float16 lB[128*32];  // [n][k]

    const int tid = threadIdx.x;
    const int l  = tid & 31, w = tid >> 5;
    const int hi = l >> 4,  r = l & 15;
    const int wm = w >> 1,  wn = w & 1;
    const int m0 = blockIdx.x * 128;
    const int n0 = blockIdx.y * 128;

    const float*    Af32 = (const float*)Ain;
    const _Float16* Af16 = (const _Float16*)Ain;

    v8f acc[2][4];
    #pragma unroll
    for (int i = 0; i < 2; ++i)
        #pragma unroll
        for (int j = 0; j < 4; ++j) acc[i][j] = v8f_zero();

    const int lrow = tid >> 1;        // 0..127
    const int lk   = (tid & 1) * 16;  // 0 or 16

    for (int kk = 0; kk < D_MODEL; kk += 32) {
        // ---- stage A tile ----
        if (A_F16) {
            // raw f16 rows: 128 rows x 64B; 256 threads x 32B via async DMA
            const _Float16* gsrc = Af16 + (size_t)(m0 + lrow) * D_MODEL + kk + lk;
            unsigned ldst = (unsigned)(uintptr_t)&lA[lrow*32 + lk];
            async_copy_b128(ldst,      gsrc);
            async_copy_b128(ldst + 16, (const char*)gsrc + 16);
        } else {
            size_t ga = (size_t)(m0 + lrow) * D_MODEL + kk + lk;
            #pragma unroll
            for (int i = 0; i < 16; ++i) lA[lrow*32 + lk + i] = (_Float16)Af32[ga + i];
        }
        // ---- stage W tile (always f32 -> f16 convert) ----
        {
            size_t gw = (size_t)(n0 + lrow) * D_MODEL + kk + lk;
            #pragma unroll
            for (int i = 0; i < 16; ++i) lB[lrow*32 + lk + i] = (_Float16)W[gw + i];
        }
        if (A_F16) wait_asynccnt0();
        __syncthreads();

        // A fragments: lane(<16) rows 0-15, K pairs {0-7,16-23}; lanes>=16: +8
        HFrag af[2];
        #pragma unroll
        for (int mt = 0; mt < 2; ++mt) {
            const unsigned* ap = (const unsigned*)&lA[(wm*32 + mt*16 + r) * 32];
            #pragma unroll
            for (int j = 0; j < 8; ++j)
                af[mt].u[j] = ap[j + ((j >= 4) ? 4 : 0) + 4*hi];
        }
        #pragma unroll
        for (int nt = 0; nt < 4; ++nt) {
            // B fragment: lane = column n, K pairs {0-7,16-23}; lanes>=16: +8
            HFrag bf;
            const unsigned* bp = (const unsigned*)&lB[(wn*64 + nt*16 + r) * 32];
            #pragma unroll
            for (int j = 0; j < 8; ++j) bf.u[j] = bp[j + ((j >= 4) ? 4 : 0) + 4*hi];
            #pragma unroll
            for (int mt = 0; mt < 2; ++mt)
                acc[mt][nt] = wmma_f16(af[mt].v, bf.v, acc[mt][nt]);
        }
        __syncthreads();
    }

    // epilogue: bias + optional RoPE + store
    #pragma unroll
    for (int mt = 0; mt < 2; ++mt) {
        #pragma unroll
        for (int nt = 0; nt < 4; ++nt) {
            #pragma unroll
            for (int g = 0; g < 8; ++g) {
                int m = m0 + wm*32 + mt*16 + g + 8*hi;
                int n = n0 + wn*64 + nt*16 + r;
                float c = acc[mt][nt][g] + bias[n];
                if (rope) {
                    // even/odd channel pairs live in adjacent lanes (n = r)
                    float partner = __shfl_xor(c, 1);
                    int spos = m & (SEQ - 1);
                    int pair = (n & (HDIM - 1)) >> 1;
                    float cz = cosT[spos * (HDIM/2) + pair];
                    float sz = sinT[spos * (HDIM/2) + pair];
                    c = ((n & 1) == 0) ? (c*cz - partner*sz) : (partner*sz + c*cz);
                }
                size_t oidx;
                if (out_layout == 0) {  // (B,H,S,Dh)
                    int b_ = m >> 11, s_ = m & (SEQ - 1);
                    int h_ = n >> 7,  d_ = n & (HDIM - 1);
                    oidx = (((size_t)b_*NHEADS + h_)*SEQ + s_)*HDIM + d_;
                } else {                // (B*S, D_MODEL)
                    oidx = (size_t)m * D_MODEL + n;
                }
                if (out32) out32[oidx] = c;
                if (out16) out16[oidx] = (_Float16)c;
            }
        }
    }
}

// ---------------------------------------------------------------------------
// Flash attention per (b,h). Block: 128 query rows, 8 waves x 16 rows.
// Streams causal key blocks of 32; K staged row-major, V transposed in LDS.
// ---------------------------------------------------------------------------
__global__ __launch_bounds__(256)
void attn_kernel(const _Float16* __restrict__ Q, const float* __restrict__ K,
                 const float* __restrict__ V, _Float16* __restrict__ ctx)
{
    __shared__ _Float16 kt[32*128];    // [key][d]
    __shared__ _Float16 vt[128*32];    // [d][key]  (transposed)
    __shared__ _Float16 pt[8][16*32];  // per-wave P staging [row][col]

    const int tid = threadIdx.x;
    const int l  = tid & 31, w = tid >> 5;
    const int hi = l >> 4,  r = l & 15;
    const int bh = blockIdx.y;
    const int b_ = bh >> 4, h_ = bh & 15;
    const int q0 = blockIdx.x * 128;

    const size_t bhOff = (size_t)bh * SEQ * HDIM;
    const float  scale = 0.08838834764831845f;  // 1/sqrt(128)

    // Q fragments for this wave's 16 rows (A layout, loaded from global f16)
    HFrag qf[4];
    {
        const _Float16* qrow = Q + bhOff + (size_t)(q0 + w*16 + r) * HDIM;
        #pragma unroll
        for (int dc = 0; dc < 4; ++dc) {
            const unsigned* qp = (const unsigned*)(qrow + dc*32);
            #pragma unroll
            for (int j = 0; j < 8; ++j)
                qf[dc].u[j] = qp[j + ((j >= 4) ? 4 : 0) + 4*hi];
        }
    }

    v8f   acc[8];
    float mstat[8], lstat[8];
    #pragma unroll
    for (int i = 0; i < 8; ++i) { acc[i] = v8f_zero(); mstat[i] = -1e30f; lstat[i] = 0.f; }

    const int nkb = q0/32 + 4;  // causal: last needed key block
    for (int kb = 0; kb < nkb; ++kb) {
        const int kbase = kb * 32;
        __syncthreads();  // protect LDS reuse from previous iteration
        {
            const int key  = tid >> 3;         // 0..31
            const int doff = (tid & 7) * 16;   // 0..112
            const float* krow = K + bhOff + (size_t)(kbase + key)*HDIM + doff;
            const float* vrow = V + bhOff + (size_t)(kbase + key)*HDIM + doff;
            #pragma unroll
            for (int i = 0; i < 16; ++i) kt[key*128 + doff + i]  = (_Float16)krow[i];
            #pragma unroll
            for (int i = 0; i < 16; ++i) vt[(doff + i)*32 + key] = (_Float16)vrow[i];
        }
        __syncthreads();

        // scores: 16 rows x 32 keys = two 16x16 C tiles, K over d in 4 chunks
        v8f st[2];
        st[0] = v8f_zero(); st[1] = v8f_zero();
        #pragma unroll
        for (int nt = 0; nt < 2; ++nt) {
            #pragma unroll
            for (int dc = 0; dc < 4; ++dc) {
                HFrag bf;  // B[k=d][n=key]: lane = key row, d pairs {0-7,16-23}; lanes>=16: +8
                const unsigned* kp = (const unsigned*)&kt[(nt*16 + r)*128 + dc*32];
                #pragma unroll
                for (int j = 0; j < 8; ++j) bf.u[j] = kp[j + ((j >= 4) ? 4 : 0) + 4*hi];
                st[nt] = wmma_f16(qf[dc].v, bf.v, st[nt]);
            }
        }

        // online softmax (row stats per C-VGPR; 16-lane half-wave reductions)
        #pragma unroll
        for (int g = 0; g < 8; ++g) {
            const int qr = q0 + w*16 + g + 8*hi;
            float s0 = st[0][g] * scale;
            float s1 = st[1][g] * scale;
            if (kbase + r      > qr) s0 = -1e30f;
            if (kbase + 16 + r > qr) s1 = -1e30f;
            float mx = fmaxf(s0, s1);
            #pragma unroll
            for (int msk = 1; msk < 16; msk <<= 1) mx = fmaxf(mx, __shfl_xor(mx, msk));
            float mnew = fmaxf(mstat[g], mx);
            float corr = __expf(mstat[g] - mnew);
            float p0 = __expf(s0 - mnew);
            float p1 = __expf(s1 - mnew);
            float rs = p0 + p1;
            #pragma unroll
            for (int msk = 1; msk < 16; msk <<= 1) rs += __shfl_xor(rs, msk);
            lstat[g] = lstat[g]*corr + rs;
            mstat[g] = mnew;
            #pragma unroll
            for (int nt = 0; nt < 8; ++nt) acc[nt][g] *= corr;
            pt[w][(g + 8*hi)*32 + r]      = (_Float16)p0;
            pt[w][(g + 8*hi)*32 + 16 + r] = (_Float16)p1;
        }

        // P(16x32 f16, A layout via LDS round-trip) x V(32x128)
        HFrag pf;
        {
            const unsigned* pp = (const unsigned*)&pt[w][r*32];
            #pragma unroll
            for (int j = 0; j < 8; ++j)
                pf.u[j] = pp[j + ((j >= 4) ? 4 : 0) + 4*hi];
        }
        #pragma unroll
        for (int nt = 0; nt < 8; ++nt) {
            HFrag vf;  // B[k=key][n=d]: lane = d column, key pairs {0-7,16-23}; lanes>=16: +8
            const unsigned* vp = (const unsigned*)&vt[(nt*16 + r)*32];
            #pragma unroll
            for (int j = 0; j < 8; ++j) vf.u[j] = vp[j + ((j >= 4) ? 4 : 0) + 4*hi];
            acc[nt] = wmma_f16(pf.v, vf.v, acc[nt]);
        }
    }

    // normalize and write context in (b, s, h*128+d) layout for the O-proj GEMM
    #pragma unroll
    for (int nt = 0; nt < 8; ++nt) {
        #pragma unroll
        for (int g = 0; g < 8; ++g) {
            int s_ = q0 + w*16 + g + 8*hi;
            int d_ = nt*16 + r;
            float o = acc[nt][g] / lstat[g];
            ctx[((size_t)b_*SEQ + s_)*D_MODEL + h_*HDIM + d_] = (_Float16)o;
        }
    }
}

// ---------------------------------------------------------------------------
extern "C" void kernel_launch(void* const* d_in, const int* in_sizes, int n_in,
                              void* d_out, int out_size, void* d_ws, size_t ws_size,
                              hipStream_t stream)
{
    const float* x    = (const float*)d_in[0];
    const float* fc   = (const float*)d_in[1];
    const float* fs   = (const float*)d_in[2];
    const float* wq_w = (const float*)d_in[3];
    const float* wq_b = (const float*)d_in[4];
    const float* wk_w = (const float*)d_in[5];
    const float* wk_b = (const float*)d_in[6];
    const float* wv_w = (const float*)d_in[7];
    const float* wv_b = (const float*)d_in[8];
    const float* wo_w = (const float*)d_in[9];
    const float* wo_b = (const float*)d_in[10];

    const size_t elems = (size_t)BATCH * NHEADS * SEQ * HDIM;  // 8.39M
    float* out_o = (float*)d_out;          // (B,S,D)
    float* out_k = out_o + elems;          // (B,H,S,Dh) fp32
    float* out_v = out_k + elems;          // (B,H,S,Dh) fp32

    _Float16* ws_q   = (_Float16*)d_ws;    // (B,H,S,Dh) f16
    _Float16* ws_ctx = ws_q + elems;       // (B,S,D)    f16

    dim3 blk(256, 1, 1);
    dim3 gg(MTOT/128, D_MODEL/128, 1);     // 32 x 16
    dim3 ga(SEQ/128, BATCH*NHEADS, 1);     // 16 x 32

    // Q = rope(x @ wq^T + bq) -> f16 workspace
    gemm_rope_kernel<0><<<gg, blk, 0, stream>>>(x, wq_w, wq_b, fc, fs, 1,
                                                nullptr, ws_q, 0);
    // K = rope(x @ wk^T + bk) -> fp32 d_out (also attention source, L2-resident)
    gemm_rope_kernel<0><<<gg, blk, 0, stream>>>(x, wk_w, wk_b, fc, fs, 1,
                                                out_k, nullptr, 0);
    // V = x @ wv^T + bv -> fp32 d_out
    gemm_rope_kernel<0><<<gg, blk, 0, stream>>>(x, wv_w, wv_b, nullptr, nullptr, 0,
                                                out_v, nullptr, 0);
    // flash attention -> f16 context
    attn_kernel<<<ga, blk, 0, stream>>>(ws_q, out_k, out_v, ws_ctx);
    // output projection (f16 A staged via async global->LDS DMA)
    gemm_rope_kernel<1><<<gg, blk, 0, stream>>>(ws_ctx, wo_w, wo_b, nullptr, nullptr, 0,
                                                out_o, nullptr, 1);
}
